// BiAttention_10651518894295
// MI455X (gfx1250) — matrix operand order repaired and checked by reading, not compiled, not run on hardware
//
#include <hip/hip_runtime.h>

typedef unsigned short u16;
typedef unsigned int   u32;
typedef unsigned long long u64;
typedef __attribute__((ext_vector_type(16))) __bf16 v16bf;
typedef __attribute__((ext_vector_type(8)))  float  v8f;
typedef __attribute__((ext_vector_type(4)))  unsigned int v4u;
typedef __attribute__((ext_vector_type(8)))  int v8i;
typedef __attribute__((ext_vector_type(4)))  int v4i;

#define NEGV (-1e9f)

#if __has_builtin(__builtin_amdgcn_tensor_load_to_lds)
#define HAVE_TDM 1
#else
#define HAVE_TDM 0
#endif

// Problem dims (fixed by reference)
static constexpr int CB  = 16;
static constexpr int CQL = 128;
static constexpr int CPL = 4096;
static constexpr int CH  = 1024;

__device__ __forceinline__ u16 f2bf(float f) {
  u32 u = __float_as_uint(f);
  u32 r = u + 0x7FFFu + ((u >> 16) & 1u);   // round-to-nearest-even
  return (u16)(r >> 16);
}

// ---- CDNA5 async global->LDS copy (ASYNCcnt-tracked) ----
__device__ __forceinline__ void async_b128(u32 lds_off, u64 gaddr) {
  asm volatile("global_load_async_to_lds_b128 %0, %1, off"
               :: "v"(lds_off), "v"(gaddr) : "memory");
}
__device__ __forceinline__ void wait_async0() {
  asm volatile("s_wait_asynccnt 0x0" ::: "memory");
}
__device__ __forceinline__ void wait_tensor0() {
#if __has_builtin(__builtin_amdgcn_s_wait_tensorcnt)
  __builtin_amdgcn_s_wait_tensorcnt(0);
#else
  asm volatile("s_wait_tensorcnt 0x0" ::: "memory");
#endif
}

#if HAVE_TDM
// ---- Tensor Data Mover: 2D tile global->LDS with HW row padding (ISA ch.8 D#) ----
// tile_w elems (data_size=2B) x tile_h rows; global row stride = stride_elems.
// LDS rows padded: (pad_amount_code+1) DWORDs inserted every (2<<pad_interval_code) DWORDs.
__device__ __forceinline__ void tdm_load_2d(u32 lds_off, const void* gaddr,
                                            u32 tile_w, u32 tile_h,
                                            u32 stride_elems,
                                            u32 pad_interval_code, u32 pad_amount_code) {
  const u64 ga = (u64)(size_t)gaddr;
  v4u g0;
  g0[0] = 1u;                                            // count=1, user mode, no gather
  g0[1] = lds_off;                                       // lds_addr (bytes)
  g0[2] = (u32)ga;                                       // global_addr[31:0]
  g0[3] = (u32)((ga >> 32) & 0x1FFFFFFu) | (2u << 30);   // global_addr[56:32] | type=2
  const u32 td = 0x100000u;                              // large tensor dims: never clip
  v8i g1;
  g1[0] = (int)((1u << 16)                               // data_size = 2 bytes
              | (1u << 20)                               // pad_enable
              | (pad_interval_code << 22)
              | (pad_amount_code << 25));
  g1[1] = (int)((td & 0xFFFFu) << 16);                   // tensor_dim0[15:0]
  g1[2] = (int)(((td >> 16) & 0xFFFFu) | ((td & 0xFFFFu) << 16));   // dim0 hi | dim1 lo
  g1[3] = (int)(((td >> 16) & 0xFFFFu) | ((tile_w & 0xFFFFu) << 16));// dim1 hi | tile_dim0
  g1[4] = (int)(tile_h & 0xFFFFu);                       // tile_dim1 (tile_dim2=0: 2D)
  g1[5] = (int)stride_elems;                             // tensor_dim0_stride[31:0]
  g1[6] = 0;                                             // stride hi | dim1_stride lo
  g1[7] = 0;
  const v4i z4 = {0, 0, 0, 0};                           // groups 2/3 unused (2D tensor)
  const v8i z8 = {0, 0, 0, 0, 0, 0, 0, 0};
  __builtin_amdgcn_tensor_load_to_lds(g0, g1, z4, z4, z8, 0);
}
#endif

// ---------------- fp32 -> bf16 conversion (vectorized) ----------------
__global__ __launch_bounds__(256) void cvt_bf16(const float4* __restrict__ in,
                                                u64* __restrict__ out, long n4) {
  long i = (long)blockIdx.x * blockDim.x + threadIdx.x;
  if (i >= n4) return;
  float4 f = in[i];
  u64 pk = (u64)f2bf(f.x) | ((u64)f2bf(f.y) << 16) |
           ((u64)f2bf(f.z) << 32) | ((u64)f2bf(f.w) << 48);
  out[i] = pk;
}

// ---------------- WMMA GEMM, TDM/async double-buffered ----------------
// C[M,N] = A[M,K] * B, batched over blockIdx.z.
//   B_IS_NK = true : B stored as N rows of K (C = A * B^T; projections / scores)
//   B_IS_NK = false: B stored as K rows of N (attended outputs)
// Block tile 128(M) x 256(N), K-step 32. 8 waves, each owns 64x64 (16 WMMAs/K-step).
union Frag { v16bf v; u32 u[8]; };

template<bool B_IS_NK, bool OUT_BF16, bool OUT_NT>
__global__ __launch_bounds__(256) void gemm_wmma(
    const u16* __restrict__ A, const u16* __restrict__ B, void* __restrict__ Cv,
    int M, int N, int K, long sA, long sB, long sC)
{
  constexpr int LSA  = 40;    // A LDS row stride (elems): [m][k], 80B rows
  constexpr int LSB  = 40;    // B NK LDS row stride: [n][k]
  constexpr int LSBT = 264;   // B NT LDS row stride: [k][n], 528B rows
  __shared__ u16 lA[2][128 * LSA];
  __shared__ u16 lB[2][B_IS_NK ? (256 * LSB) : (32 * LSBT)];

  const int bz = blockIdx.z;
  const u16* Ab = A + (long)bz * sA;
  const u16* Bb = B + (long)bz * sB;
  const int n0 = blockIdx.x * 256;
  const int m0 = blockIdx.y * 128;
  const int t  = threadIdx.x;
  const int w  = t >> 5;                 // wave id 0..7
  const int l  = t & 31;
  const int half   = l >> 4;
  const int lane16 = l & 15;
  const int wm = (w & 1) * 64;           // wave M offset: 4 subtiles
  const int wn = (w >> 1) * 64;          // wave N offset: 4 subtiles

  v8f acc[4][4];
  #pragma unroll
  for (int mi = 0; mi < 4; ++mi)
    #pragma unroll
    for (int ni = 0; ni < 4; ++ni)
      #pragma unroll
      for (int e = 0; e < 8; ++e) acc[mi][ni][e] = 0.0f;

  // Issue one K-tile of global->LDS copies into buffer `buf`.
#if HAVE_TDM
  // One TDM descriptor per tile, issued by wave 0 only (EXEC-ignored DMA).
  auto issue = [&](int buf, int k0) {
    tdm_load_2d((u32)(size_t)&lA[buf][0], Ab + (long)m0 * K + k0,
                32, 128, (u32)K, /*16 DW*/3, /*4 DW*/3);
    if constexpr (B_IS_NK)
      tdm_load_2d((u32)(size_t)&lB[buf][0], Bb + (long)n0 * K + k0,
                  32, 256, (u32)K, 3, 3);
    else
      tdm_load_2d((u32)(size_t)&lB[buf][0], Bb + (long)k0 * N + n0,
                  256, 32, (u32)N, /*128 DW*/6, /*4 DW*/3);
  };
#else
  auto issue = [&](int buf, int k0) {
    {   // A tile: 128 x 32, contiguous per row
      const int ar = t >> 1, ac = (t & 1) << 4;
      const u16* g = Ab + (long)(m0 + ar) * K + k0 + ac;
      const u32 lofs = (u32)(size_t)&lA[buf][ar * LSA + ac];
      async_b128(lofs,      (u64)(size_t)g);
      async_b128(lofs + 16, (u64)(size_t)(g + 8));
    }
    if constexpr (B_IS_NK) {            // B tile: 256 rows x 32 k
      const u16* g = Bb + (long)(n0 + t) * K + k0;
      const u32 lofs = (u32)(size_t)&lB[buf][t * LSB];
      #pragma unroll
      for (int j = 0; j < 4; ++j)
        async_b128(lofs + 16 * j, (u64)(size_t)(g + 8 * j));
    } else {                            // B tile: 32 k-rows x 256 n, kept [k][n]
      const int kk = t >> 3, c = (t & 7) << 5;
      const u16* g = Bb + (long)(k0 + kk) * N + n0 + c;
      const u32 lofs = (u32)(size_t)&lB[buf][kk * LSBT + c];
      #pragma unroll
      for (int j = 0; j < 4; ++j)
        async_b128(lofs + 16 * j, (u64)(size_t)(g + 8 * j));
    }
  };
#endif

#if HAVE_TDM
  if (w == 0) issue(0, 0);
#else
  issue(0, 0);
#endif
  int cur = 0;

  for (int k0 = 0; k0 < K; k0 += 32) {
#if HAVE_TDM
    if (w == 0) wait_tensor0();
#else
    wait_async0();
#endif
    __syncthreads();                    // buf[cur] fully landed in LDS, all waves
    if (k0 + 32 < K) {                  // overlap next tile DMA with compute
#if HAVE_TDM
      if (w == 0) issue(cur ^ 1, k0 + 32);
#else
      issue(cur ^ 1, k0 + 32);
#endif
    }

    // ---- A fragments (contiguous k = 16*half + s, same order as B) ----
    Frag af[4];
    #pragma unroll
    for (int mi = 0; mi < 4; ++mi) {
      const int base = (wm + mi * 16 + lane16) * LSA;
      #pragma unroll
      for (int v = 0; v < 8; ++v) {
        const int k = half * 16 + (v << 1);
        af[mi].u[v] = *reinterpret_cast<const u32*>(&lA[cur][base + k]);
      }
    }
    // ---- B fragments ----
    Frag bfr[4];
    if constexpr (B_IS_NK) {
      #pragma unroll
      for (int ni = 0; ni < 4; ++ni) {
        const int base = (wn + ni * 16 + lane16) * LSB;
        #pragma unroll
        for (int v = 0; v < 8; ++v) {
          const int k = half * 16 + (v << 1);
          bfr[ni].u[v] = *reinterpret_cast<const u32*>(&lB[cur][base + k]);
        }
      }
    } else {
      // gather-transpose from [k][n]: pack (k, k+1) u16 pair into one dword
      #pragma unroll
      for (int ni = 0; ni < 4; ++ni) {
        const int n = wn + ni * 16 + lane16;
        #pragma unroll
        for (int v = 0; v < 8; ++v) {
          const int k = half * 16 + (v << 1);
          const u32 lo = lB[cur][k * LSBT + n];
          const u32 hi = lB[cur][(k + 1) * LSBT + n];
          bfr[ni].u[v] = lo | (hi << 16);
        }
      }
    }

    // ---- 16 WMMAs per wave per K-step ----
    #pragma unroll
    for (int mi = 0; mi < 4; ++mi)
      #pragma unroll
      for (int ni = 0; ni < 4; ++ni)
        acc[mi][ni] = __builtin_amdgcn_wmma_f32_16x16x32_bf16(
            false, af[mi].v, false, bfr[ni].v, (short)0, acc[mi][ni], false, false);

    __syncthreads();                    // compute done before buf[cur] is refilled
    cur ^= 1;
  }

  // ---- epilogue: D layout M = v + half*8, N = lane16 ----
  #pragma unroll
  for (int mi = 0; mi < 4; ++mi) {
    #pragma unroll
    for (int ni = 0; ni < 4; ++ni) {
      const int n = n0 + wn + ni * 16 + lane16;
      #pragma unroll
      for (int v = 0; v < 8; ++v) {
        const int m = m0 + wm + mi * 16 + half * 8 + v;
        const long idx = (long)bz * sC + (long)m * N + n;
        if constexpr (OUT_BF16) {
          reinterpret_cast<u16*>(Cv)[idx] = f2bf(acc[mi][ni][v]);
        } else if constexpr (OUT_NT) {
          __builtin_nontemporal_store(acc[mi][ni][v], reinterpret_cast<float*>(Cv) + idx);
        } else {
          reinterpret_cast<float*>(Cv)[idx] = acc[mi][ni][v];
        }
      }
    }
  }
}

// ---------------- softmax over p (axis=2): attnP[b,q,p], bf16 ----------------
__global__ __launch_bounds__(256) void softmax_rows(
    const float* __restrict__ scores, const int* __restrict__ qmask,
    const int* __restrict__ pmask, u16* __restrict__ attnP)
{
  const int bq = blockIdx.x;             // b*QL + q
  const int b  = bq / CQL;
  const int q  = bq % CQL;
  const int t  = threadIdx.x;
  const float* row = scores + (long)bq * CPL;
  const int qm = qmask[b * CQL + q];
  const int* pm = pmask + b * CPL;

  __shared__ float red[256];
  float vals[CPL / 256];
  float mx = -3.4e38f;
  #pragma unroll
  for (int i = 0; i < CPL / 256; ++i) {
    const int p = i * 256 + t;
    const float v = (qm != 0 && pm[p] != 0) ? row[p] : NEGV;
    vals[i] = v;
    mx = fmaxf(mx, v);
  }
  red[t] = mx; __syncthreads();
  for (int s = 128; s > 0; s >>= 1) {
    if (t < s) red[t] = fmaxf(red[t], red[t + s]);
    __syncthreads();
  }
  mx = red[0]; __syncthreads();

  float sum = 0.0f;
  #pragma unroll
  for (int i = 0; i < CPL / 256; ++i) {
    const float e = __expf(vals[i] - mx);
    vals[i] = e; sum += e;
  }
  red[t] = sum; __syncthreads();
  for (int s = 128; s > 0; s >>= 1) {
    if (t < s) red[t] += red[t + s];
    __syncthreads();
  }
  const float inv = 1.0f / red[0];
  #pragma unroll
  for (int i = 0; i < CPL / 256; ++i)
    attnP[(long)bq * CPL + i * 256 + t] = f2bf(vals[i] * inv);
}

// ---------------- softmax over q (axis=1), transposed out: attnQT[b,p,q], bf16 ----------------
__global__ __launch_bounds__(128) void softmax_cols(
    const float* __restrict__ scores, const int* __restrict__ qmask,
    const int* __restrict__ pmask, u16* __restrict__ attnQT)
{
  const int bp = blockIdx.x;             // b*PL + p
  const int b  = bp / CPL;
  const int p  = bp % CPL;
  const int t  = threadIdx.x;            // q index, 0..127
  const float v = (qmask[b * CQL + t] != 0 && pmask[b * CPL + p] != 0)
                      ? scores[((long)b * CQL + t) * CPL + p] : NEGV;

  __shared__ float red[128];
  red[t] = v; __syncthreads();
  for (int s = 64; s > 0; s >>= 1) {
    if (t < s) red[t] = fmaxf(red[t], red[t + s]);
    __syncthreads();
  }
  const float mx = red[0]; __syncthreads();
  const float e = __expf(v - mx);
  red[t] = e; __syncthreads();
  for (int s = 64; s > 0; s >>= 1) {
    if (t < s) red[t] += red[t + s];
    __syncthreads();
  }
  attnQT[((long)b * CPL + p) * CQL + t] = f2bf(e / red[0]);
}

// ---------------- host launcher ----------------
extern "C" void kernel_launch(void* const* d_in, const int* in_sizes, int n_in,
                              void* d_out, int out_size, void* d_ws, size_t ws_size,
                              hipStream_t stream) {
  (void)in_sizes; (void)n_in; (void)out_size;
  const float* qh   = (const float*)d_in[0];
  const float* ph   = (const float*)d_in[1];
  const int* qmask  = (const int*)d_in[2];
  const int* pmask  = (const int*)d_in[3];
  const float* wq   = (const float*)d_in[4];
  const float* wp   = (const float*)d_in[5];

  char* ws = (char*)d_ws;
  size_t off = 0;
  auto alloc = [&](size_t bytes) -> void* {
    void* p = ws + off;
    off += (bytes + 255) & ~((size_t)255);
    return p;
  };
  const long nQH = (long)CB * CQL * CH;      // 2,097,152
  const long nPH = (long)CB * CPL * CH;      // 67,108,864
  const long nW  = (long)CH * CH;            // 1,048,576
  const long nSC = (long)CB * CQL * CPL;     // 8,388,608

  u16*   qh16   = (u16*)alloc(nQH * 2);
  u16*   ph16   = (u16*)alloc(nPH * 2);
  u16*   wq16   = (u16*)alloc(nW * 2);
  u16*   wp16   = (u16*)alloc(nW * 2);
  u16*   qp16   = (u16*)alloc(nQH * 2);
  u16*   pp16   = (u16*)alloc(nPH * 2);
  float* scoresF= (float*)alloc(nSC * 4);
  u16*   attnP  = (u16*)alloc(nSC * 2);
  u16*   attnQT = (u16*)alloc(nSC * 2);
  if (off > ws_size) return;                 // workspace too small: bail deterministically

  const dim3 blk(256);

  // 1) convert fp32 -> bf16
  cvt_bf16<<<(unsigned)(nQH / 1024), blk, 0, stream>>>((const float4*)qh, (u64*)qh16, nQH / 4);
  cvt_bf16<<<(unsigned)(nPH / 1024), blk, 0, stream>>>((const float4*)ph, (u64*)ph16, nPH / 4);
  cvt_bf16<<<(unsigned)(nW  / 1024), blk, 0, stream>>>((const float4*)wq, (u64*)wq16, nW / 4);
  cvt_bf16<<<(unsigned)(nW  / 1024), blk, 0, stream>>>((const float4*)wp, (u64*)wp16, nW / 4);

  // 2) projections: q_proj = qh * wq^T, p_proj = ph * wp^T (bf16 out)
  gemm_wmma<true, true, false><<<dim3(CH / 256, (CB * CQL) / 128, 1), blk, 0, stream>>>(
      qh16, wq16, qp16, CB * CQL, CH, CH, 0, 0, 0);
  gemm_wmma<true, true, false><<<dim3(CH / 256, (CB * CPL) / 128, 1), blk, 0, stream>>>(
      ph16, wp16, pp16, CB * CPL, CH, CH, 0, 0, 0);

  // 3) scores[b] = q_proj[b] * p_proj[b]^T (fp32 out, re-read by softmaxes: RT)
  gemm_wmma<true, false, false><<<dim3(CPL / 256, CQL / 128, CB), blk, 0, stream>>>(
      qp16, pp16, scoresF, CQL, CPL, CH,
      (long)CQL * CH, (long)CPL * CH, (long)CQL * CPL);

  // 4) masked softmaxes
  softmax_rows<<<CB * CQL, blk, 0, stream>>>(scoresF, qmask, pmask, attnP);
  softmax_cols<<<CB * CPL, dim3(128), 0, stream>>>(scoresF, qmask, pmask, attnQT);

  // 5) attended outputs (fp32 into d_out, streamed: non-temporal stores)
  float* out_q2p = (float*)d_out;                       // (B, PL, H)
  float* out_p2q = out_q2p + (long)CB * CPL * CH;       // (B, QL, H)
  gemm_wmma<false, false, true><<<dim3(CH / 256, CPL / 128, CB), blk, 0, stream>>>(
      attnQT, qh16, out_q2p, CPL, CH, CQL,
      (long)CPL * CQL, (long)CQL * CH, (long)CPL * CH);
  gemm_wmma<false, false, true><<<dim3(CH / 256, CQL / 128, CB), blk, 0, stream>>>(
      attnP, ph16, out_p2q, CQL, CH, CPL,
      (long)CQL * CPL, (long)CPL * CH, (long)CQL * CH);
}
